// Attention_44968307589689
// MI455X (gfx1250) — matrix operand order repaired and checked by reading, not compiled, not run on hardware
//
#include <hip/hip_runtime.h>
#include <hip/hip_bf16.h>
#include <math.h>
#include <stdint.h>

typedef __bf16 bf16;
typedef __attribute__((ext_vector_type(16))) __bf16 v16bf;
typedef __attribute__((ext_vector_type(8)))  __bf16 v8bf;
typedef __attribute__((ext_vector_type(8)))  float  v8f;

// ---------------------------------------------------------------------------
// WMMA helper (CDNA5 gfx1250, wave32): D(16x16,f32) = A(16x32,bf16) B(32x16,bf16) + C
// ---------------------------------------------------------------------------
__device__ __forceinline__ v8f wmma_bf16(v16bf a, v16bf b, v8f c) {
  return __builtin_amdgcn_wmma_f32_16x16x32_bf16(
      false, a, false, b, (short)0, c, false, false);
}

// A-fragment: A[M,K] row-major bf16 (contiguous k order, matches B):
//   lanes 0-15 : row=lane,    h0..15 -> K+0..15
//   lanes16-31 : row=lane-16, h0..15 -> K+16..31
__device__ __forceinline__ v16bf load_a_frag(const bf16* A, int ld,
                                             int m_base, int k_base, int lane) {
  int hi = lane >> 4;
  int r  = lane & 15;
  const bf16* row = A + (size_t)(m_base + r) * ld + k_base;
  v8bf c0 = *(const v8bf*)(row + hi * 16);
  v8bf c1 = *(const v8bf*)(row + hi * 16 + 8);
  v16bf a;
#pragma unroll
  for (int i = 0; i < 8; ++i) { a[i] = c0[i]; a[8 + i] = c1[i]; }
  return a;
}

// B-fragment from weight-style W[N,K] row-major (B = W^T, 32x16):
//   lanes 0-15 : col=lane,    K offs 0..15  -> W[n][k..k+15] (one 32B load)
//   lanes16-31 : col=lane-16, K offs 16..31 -> W[n][k+16..k+31]
__device__ __forceinline__ v16bf load_b_weight(const bf16* W, int ld,
                                               int n_base, int k_base, int lane) {
  int hi = lane >> 4;
  int n  = lane & 15;
  return *(const v16bf*)(W + (size_t)(n_base + n) * ld + k_base + hi * 16);
}

// ---------------------------------------------------------------------------
// f32 -> bf16 conversion
// ---------------------------------------------------------------------------
__global__ void f32_to_bf16_kernel(const float* __restrict__ in,
                                   bf16* __restrict__ out, int n) {
  int i = blockIdx.x * blockDim.x + threadIdx.x;
  if (i < n) out[i] = (bf16)in[i];
}

// ---------------------------------------------------------------------------
// Transpose V slice of qkv into vt[b][d][s]
// ---------------------------------------------------------------------------
__global__ void transpose_v_kernel(const bf16* __restrict__ qkv,
                                   bf16* __restrict__ vt) {
  int i = blockIdx.x * blockDim.x + threadIdx.x;   // over 16384*768
  if (i >= 16384 * 768) return;
  int m = i / 768, d = i % 768;
  int b = m >> 10, s = m & 1023;
  vt[(size_t)b * 768 * 1024 + (size_t)d * 1024 + s] =
      qkv[(size_t)m * 2304 + 1536 + d];
}

// ---------------------------------------------------------------------------
// GEMM: out[M,N] = A[M,K](bf16) @ W[N,K]^T(bf16) + bias[N](f32)
// grid.x = M/16; wave handles 2 consecutive N-tiles (A-frag reused 2x).
// ---------------------------------------------------------------------------
template <typename OutT>
__global__ void gemm_bias_kernel(const bf16* __restrict__ A,
                                 const bf16* __restrict__ W,
                                 const float* __restrict__ bias,
                                 OutT* __restrict__ out, int N, int K) {
  int lane   = threadIdx.x & 31;
  int wave   = threadIdx.x >> 5;
  int m_base = blockIdx.x * 16;
  int n_base = (blockIdx.y * 8 + wave) * 32;

  v8f c0 = {}, c1 = {};
  for (int kk = 0; kk < K; kk += 32) {
    v16bf a  = load_a_frag(A, K, m_base, kk, lane);
    v16bf b0 = load_b_weight(W, K, n_base, kk, lane);
    v16bf b1 = load_b_weight(W, K, n_base + 16, kk, lane);
    c0 = wmma_bf16(a, b0, c0);
    c1 = wmma_bf16(a, b1, c1);
  }

  int n  = lane & 15;
  int mo = (lane >> 4) * 8;
  float bi0 = bias[n_base + n];
  float bi1 = bias[n_base + 16 + n];
#pragma unroll
  for (int i = 0; i < 8; ++i) {
    size_t row = (size_t)(m_base + mo + i);
    out[row * N + n_base + n]      = (OutT)(c0[i] + bi0);
    out[row * N + n_base + 16 + n] = (OutT)(c1[i] + bi1);
  }
}

// ---------------------------------------------------------------------------
// Fused attention: one workgroup = one (batch, 16-row q tile). 8 waves, 96KB LDS.
// Q tile staged to LDS via gfx1250 async copy (ASYNCcnt); Pm reused as staging.
// ---------------------------------------------------------------------------
__global__ void attn_kernel(const bf16* __restrict__ qkv,
                            const bf16* __restrict__ vt,
                            bf16* __restrict__ attn_out, float scale) {
  __shared__ __align__(16) float Ssc[16 * 1024];   // 64 KB
  __shared__ __align__(16) bf16  Pm[16 * 1024];    // 32 KB (also Q staging)
  __shared__ float red[16][16];
  __shared__ float rmax[16], rinv[16];

  int tile   = blockIdx.x;        // 0..1023
  int b      = tile >> 6;
  int q_base = (tile & 63) * 16;

  const bf16* Qb  = qkv + (size_t)b * 1024 * 2304;
  const bf16* Kb  = Qb + 768;
  const bf16* Vtb = vt + (size_t)b * 768 * 1024;

  int lane = threadIdx.x & 31;
  int wave = threadIdx.x >> 5;

  // ---- Stage Q tile (16 x 768 bf16 = 24KB) into LDS (Pm) asynchronously ----
  bf16* Qs = Pm;
  {
    uint32_t lds_base = (uint32_t)(uintptr_t)Qs;            // wg-relative LDS addr
    const char* gbase = (const char*)(Qb + (size_t)q_base * 2304);
    // 1536 chunks of 16B; 256 threads x 6 chunks, uniform trip count
#pragma unroll
    for (int it = 0; it < 6; ++it) {
      int c = it * 256 + threadIdx.x;
      int row   = c / 96;
      int off   = (c % 96) * 16;                            // byte offset in row
      uint64_t gaddr = (uint64_t)(uintptr_t)(gbase + (size_t)row * 4608 + off);
      uint32_t laddr = lds_base + row * 1536 + off;
      asm volatile("global_load_async_to_lds_b128 %0, %1, off"
                   :: "v"(laddr), "v"(gaddr) : "memory");
    }
    asm volatile("s_wait_asynccnt 0" ::: "memory");
  }
  __syncthreads();

  // ---- Phase 1: S = scale * Q K^T (wave covers keys [wave*128, +128)) ----
  {
    v8f acc[8];
#pragma unroll
    for (int t = 0; t < 8; ++t) acc[t] = (v8f){};
    int key0 = wave * 128;
    for (int kk = 0; kk < 768; kk += 32) {
      v16bf a = load_a_frag(Qs, 768, 0, kk, lane);          // from LDS
#pragma unroll
      for (int t = 0; t < 8; t += 2) {                      // paired B frags
        v16bf bk0 = load_b_weight(Kb, 2304, key0 + t * 16, kk, lane);
        v16bf bk1 = load_b_weight(Kb, 2304, key0 + (t + 1) * 16, kk, lane);
        acc[t]     = wmma_bf16(a, bk0, acc[t]);
        acc[t + 1] = wmma_bf16(a, bk1, acc[t + 1]);
      }
    }
    int n  = lane & 15;
    int mo = (lane >> 4) * 8;
#pragma unroll
    for (int t = 0; t < 8; ++t)
#pragma unroll
      for (int i = 0; i < 8; ++i)
        Ssc[(mo + i) * 1024 + key0 + t * 16 + n] = acc[t][i] * scale;
  }
  __syncthreads();

  // ---- Phase 2: softmax (16 threads per row) -> Pm (overwrites Q staging) ----
  {
    int r = threadIdx.x >> 4;
    int p = threadIdx.x & 15;
    const float* row = Ssc + r * 1024;
    float m = -1e30f;
    for (int j = p * 64; j < p * 64 + 64; ++j) m = fmaxf(m, row[j]);
    red[r][p] = m;
    __syncthreads();
    if (p == 0) {
      float mm = red[r][0];
#pragma unroll
      for (int t = 1; t < 16; ++t) mm = fmaxf(mm, red[r][t]);
      rmax[r] = mm;
    }
    __syncthreads();
    float mm = rmax[r];
    float s = 0.0f;
    for (int j = p * 64; j < p * 64 + 64; ++j) s += __expf(row[j] - mm);
    red[r][p] = s;
    __syncthreads();
    if (p == 0) {
      float ss = 0.0f;
#pragma unroll
      for (int t = 0; t < 16; ++t) ss += red[r][t];
      rinv[r] = 1.0f / ss;
    }
    __syncthreads();
    for (int i = threadIdx.x; i < 16 * 1024; i += 256) {
      int rr = i >> 10;
      Pm[i] = (bf16)(__expf(Ssc[i] - rmax[rr]) * rinv[rr]);
    }
  }
  __syncthreads();

  // ---- Phase 3: O = P @ V (wave owns 6 head-dim tiles = 96 cols) ----
  {
    v8f acc[6];
#pragma unroll
    for (int j = 0; j < 6; ++j) acc[j] = (v8f){};
    int n0 = wave * 96;
    for (int kk = 0; kk < 1024; kk += 32) {
      v16bf a = load_a_frag(Pm, 1024, 0, kk, lane);
#pragma unroll
      for (int j = 0; j < 6; j += 2) {                      // paired B frags
        v16bf bv0 = load_b_weight(Vtb, 1024, n0 + j * 16, kk, lane);
        v16bf bv1 = load_b_weight(Vtb, 1024, n0 + (j + 1) * 16, kk, lane);
        acc[j]     = wmma_bf16(a, bv0, acc[j]);
        acc[j + 1] = wmma_bf16(a, bv1, acc[j + 1]);
      }
    }
    int n  = lane & 15;
    int mo = (lane >> 4) * 8;
#pragma unroll
    for (int j = 0; j < 6; ++j)
#pragma unroll
      for (int i = 0; i < 8; ++i)
        attn_out[(size_t)(b * 1024 + q_base + mo + i) * 768 + n0 + j * 16 + n] =
            (bf16)acc[j][i];
  }
}

// ---------------------------------------------------------------------------
// Launch
// ---------------------------------------------------------------------------
static inline size_t align_up(size_t v, size_t a) { return (v + a - 1) & ~(a - 1); }

extern "C" void kernel_launch(void* const* d_in, const int* in_sizes, int n_in,
                              void* d_out, int out_size, void* d_ws, size_t ws_size,
                              hipStream_t stream) {
  (void)in_sizes; (void)n_in; (void)out_size; (void)ws_size;

  const int B = 16, S = 1024, D = 768;
  const int M = B * S;            // 16384
  const int E = 3 * D;            // 2304

  const float* x     = (const float*)d_in[0];
  const float* w_qkv = (const float*)d_in[1];
  const float* b_qkv = (const float*)d_in[2];
  const float* w_out = (const float*)d_in[3];
  const float* b_out = (const float*)d_in[4];
  float* out = (float*)d_out;

  char* ws = (char*)d_ws;
  size_t off = 0;
  bf16* xh    = (bf16*)(ws + off); off = align_up(off + (size_t)M * D * 2, 256);
  bf16* wqkvh = (bf16*)(ws + off); off = align_up(off + (size_t)E * D * 2, 256);
  bf16* wouth = (bf16*)(ws + off); off = align_up(off + (size_t)D * D * 2, 256);
  bf16* qkvh  = (bf16*)(ws + off); off = align_up(off + (size_t)M * E * 2, 256);
  bf16* vth   = (bf16*)(ws + off); off = align_up(off + (size_t)M * D * 2, 256);
  bf16* attnh = (bf16*)(ws + off); off = align_up(off + (size_t)M * D * 2, 256);

  // 1) f32 -> bf16 conversions
  {
    int n = M * D;
    f32_to_bf16_kernel<<<(n + 255) / 256, 256, 0, stream>>>(x, xh, n);
    n = E * D;
    f32_to_bf16_kernel<<<(n + 255) / 256, 256, 0, stream>>>(w_qkv, wqkvh, n);
    n = D * D;
    f32_to_bf16_kernel<<<(n + 255) / 256, 256, 0, stream>>>(w_out, wouth, n);
  }

  // 2) qkv = x @ w_qkv^T + b_qkv  -> bf16 [M, 2304]
  gemm_bias_kernel<bf16><<<dim3(M / 16, E / 256), 256, 0, stream>>>(
      xh, wqkvh, b_qkv, qkvh, E, D);

  // 2b) transpose V slice -> vt[b][d][s]
  {
    int n = M * D;
    transpose_v_kernel<<<(n + 255) / 256, 256, 0, stream>>>(qkvh, vth);
  }

  // 3) fused attention -> bf16 [M, 768]
  float scale = 1.0f / sqrtf((float)D);
  attn_kernel<<<M / 16, 256, 0, stream>>>(qkvh, vth, attnh, scale);

  // 4) out = attn @ w_out^T + b_out -> f32 d_out
  gemm_bias_kernel<float><<<dim3(M / 16, D / 256), 256, 0, stream>>>(
      attnh, wouth, b_out, out, D, D);
}
